// BidirectionalAttention2_64390149702332
// MI455X (gfx1250) — matrix operand order repaired and checked by reading, not compiled, not run on hardware
//
#include <hip/hip_runtime.h>
#include <hip/hip_bf16.h>

typedef __attribute__((ext_vector_type(16))) __bf16        v16bf;
typedef __attribute__((ext_vector_type(8)))  float         v8f;
typedef __attribute__((ext_vector_type(8)))  unsigned int  v8u;
typedef __attribute__((ext_vector_type(4)))  int           v4i_t;

#define MASK_FILL (-1e-07f)

// granule-swizzled indices (ushort index of a 16B granule) -- used both for the
// precomputed global layout and the LDS tiles (identical granule order).
//   K tile  : granule(c,s,key) = 8 bf16 dims [c*32+s*8 .. +7] of one key row
//   V^T tile: granule(s,d)     = 8 bf16 keys [s*8 .. +7] of one dim row
#define KIDX(c, s, r) (((((c) * 4 + (s)) * 32) + (r)) * 8)
#define VIDX(s, dr)   ((((s) * 256 + (dr))) * 8)

#if defined(__AMDGCN__) && __has_builtin(__builtin_amdgcn_global_load_async_to_lds_b128) && \
    __has_builtin(__builtin_amdgcn_s_wait_asynccnt)
#define USE_ASYNC_LDS 1
#else
#define USE_ASYNC_LDS 0
#endif

#if USE_ASYNC_LDS
__device__ __forceinline__ void async_cp16(const void* g, void* l) {
    __builtin_amdgcn_global_load_async_to_lds_b128(
        (__attribute__((address_space(1))) v4i_t*)(unsigned long long)g,
        (__attribute__((address_space(3))) v4i_t*)(unsigned)(unsigned long long)l,
        0, 0);
}
#endif

// f32 pair -> packed bf16 dword, round-half-up via +0x8000 then byte-perm
__device__ __forceinline__ unsigned bf2(float a, float b) {
    unsigned ua = __float_as_uint(a) + 0x8000u;
    unsigned ub = __float_as_uint(b) + 0x8000u;
    return __builtin_amdgcn_perm(ub, ua, 0x07060302u);
}

// swap data between lane q and lane q+16 (SWAPX16 ds_swizzle pattern)
__device__ __forceinline__ float swapx16f(float x) {
    return __int_as_float(__builtin_amdgcn_ds_swizzle(__float_as_int(x), 0x401f));
}
__device__ __forceinline__ unsigned swapx16u(unsigned x) {
    return (unsigned)__builtin_amdgcn_ds_swizzle((int)x, 0x401f);
}

__device__ __forceinline__ unsigned ballot32(bool p) {
#if __has_builtin(__builtin_amdgcn_ballot_w32)
    return __builtin_amdgcn_ballot_w32(p);
#else
    return (unsigned)(__ballot(p) & 0xffffffffull);
#endif
}

// =====================================================================
// Pre-pass: fp32 [B][L][D] -> bf16 granule layouts (per 32-key tile):
//   Kgr [b][t][1024 granules]  (KIDX order)   VTgr [b][t][1024 granules] (VIDX order)
// =====================================================================
__global__ __launch_bounds__(256)
void pack_bf16_kernel(const float* __restrict__ X,
                      unsigned short* __restrict__ Kgr,
                      unsigned short* __restrict__ VTgr)
{
    constexpr int L = 1024, D = 256, NT = L / 32;
    const int t = blockIdx.x, b = blockIdx.y, tid = threadIdx.x;
    const int k0   = (tid & 15) * 2;
    const int dgrp = tid >> 4;
    const int d0   = dgrp * 16;
    const int c0   = dgrp >> 1;
    const int s0   = (dgrp & 1) * 2;

    const float* Xb = X + (size_t)b * L * D + (size_t)t * 32 * D;
    unsigned short* Kt = Kgr  + ((size_t)b * NT + t) * 8192;
    unsigned short* Vt = VTgr + ((size_t)b * NT + t) * 8192;

    float4 fa[4], fb[4];
    const float4* sA = (const float4*)(Xb + (size_t)k0 * D + d0);
    const float4* sB = (const float4*)(Xb + (size_t)(k0 + 1) * D + d0);
    #pragma unroll
    for (int j = 0; j < 4; ++j) { fa[j] = sA[j]; fb[j] = sB[j]; }

    unsigned kA[8], kB[8];
    #pragma unroll
    for (int j = 0; j < 4; ++j) {
        kA[2 * j]     = bf2(fa[j].x, fa[j].y);
        kA[2 * j + 1] = bf2(fa[j].z, fa[j].w);
        kB[2 * j]     = bf2(fb[j].x, fb[j].y);
        kB[2 * j + 1] = bf2(fb[j].z, fb[j].w);
    }
    *(uint4*)&Kt[KIDX(c0, s0,     k0)]     = make_uint4(kA[0], kA[1], kA[2], kA[3]);
    *(uint4*)&Kt[KIDX(c0, s0 + 1, k0)]     = make_uint4(kA[4], kA[5], kA[6], kA[7]);
    *(uint4*)&Kt[KIDX(c0, s0,     k0 + 1)] = make_uint4(kB[0], kB[1], kB[2], kB[3]);
    *(uint4*)&Kt[KIDX(c0, s0 + 1, k0 + 1)] = make_uint4(kB[4], kB[5], kB[6], kB[7]);
    const int sv = k0 >> 3, ko = k0 & 7;
    #pragma unroll
    for (int i = 0; i < 8; ++i) {
        const int d = d0 + 2 * i;
        *(unsigned*)&Vt[VIDX(sv, d)     + ko] = __builtin_amdgcn_perm(kB[i], kA[i], 0x05040100u);
        *(unsigned*)&Vt[VIDX(sv, d + 1) + ko] = __builtin_amdgcn_perm(kB[i], kA[i], 0x07060302u);
    }
}

// =====================================================================
// Fast path: flash attention consuming pre-packed bf16 granule tiles.
// =====================================================================
__global__ __launch_bounds__(256)
void fa_pre_kernel(const float* __restrict__ Q,
                   const unsigned short* __restrict__ Kgr,
                   const unsigned short* __restrict__ VTgr,
                   const unsigned char* __restrict__ qmask,
                   const unsigned char* __restrict__ kmask,
                   float* __restrict__ Out)
{
    constexpr int L = 1024, D = 256, NT = L / 32;
    const int b    = blockIdx.y;
    const int qt   = blockIdx.x;
    const int tid  = threadIdx.x;
    const int lane = tid & 31;
    const int wave = tid >> 5;
    const int hl   = lane >> 4;
    const int q16  = lane & 15;

    const float*          Qb    = Q    + (size_t)b * L * D;
    const unsigned short* KgrB  = Kgr  + (size_t)b * NT * 8192;
    const unsigned short* VTgrB = VTgr + (size_t)b * NT * 8192;
    const unsigned char*  qmb   = qmask + (size_t)b * L;
    const unsigned char*  kmb   = kmask + (size_t)b * L;
    float*                Ob    = Out + (size_t)b * L * D;

    __shared__ __align__(16) unsigned short ldsK [2][8192];
    __shared__ __align__(16) unsigned short ldsVT[2][8192];

    const int qrow = qt * 128 + wave * 16 + q16;

    // fragment loaders (granule-consecutive -> conflict-free ds_load_b128)
    auto ldKfrag = [&](int bu, int c, int t2, v8u& u) {
        uint4 x = *(const uint4*)&ldsK[bu][KIDX(c, hl,     t2 * 16 + q16)];
        uint4 y = *(const uint4*)&ldsK[bu][KIDX(c, 2 + hl, t2 * 16 + q16)];
        u[0]=x.x; u[1]=x.y; u[2]=x.z; u[3]=x.w;
        u[4]=y.x; u[5]=y.y; u[6]=y.z; u[7]=y.w;
    };
    auto ldVfrag = [&](int bu, int cd, v8u& u) {
        uint4 x = *(const uint4*)&ldsVT[bu][VIDX(hl,     cd * 16 + q16)];
        uint4 y = *(const uint4*)&ldsVT[bu][VIDX(2 + hl, cd * 16 + q16)];
        u[0]=x.x; u[1]=x.y; u[2]=x.z; u[3]=x.w;
        u[4]=y.x; u[5]=y.y; u[6]=y.z; u[7]=y.w;
    };

    // ---- Q resident as WMMA B-fragments (lane holds its query's dims, bf16)
    v16bf qf[8];
    #pragma unroll
    for (int c = 0; c < 8; ++c) {
        const float4* p = (const float4*)(Qb + (size_t)qrow * D + c * 32 + hl * 8);
        float4 f0 = p[0], f1 = p[1], f2 = p[4], f3 = p[5];
        v8u u;
        u[0] = bf2(f0.x, f0.y); u[1] = bf2(f0.z, f0.w);
        u[2] = bf2(f1.x, f1.y); u[3] = bf2(f1.z, f1.w);
        u[4] = bf2(f2.x, f2.y); u[5] = bf2(f2.z, f2.w);
        u[6] = bf2(f3.x, f3.y); u[7] = bf2(f3.z, f3.w);
        qf[c] = __builtin_bit_cast(v16bf, u);
    }

    v8f acc[16];
    { v8f z = {};
      #pragma unroll
      for (int i = 0; i < 16; ++i) acc[i] = z; }
    float mrun = -3.0e38f, lrun = 0.0f;

#if USE_ASYNC_LDS
    auto stage = [&](int nb2, int tile) {   // 8 async 16B copies per thread (4 K + 4 VT)
        const unsigned short* gk = KgrB  + (size_t)tile * 8192 + tid * 32;
        const unsigned short* gv = VTgrB + (size_t)tile * 8192 + tid * 32;
        unsigned short* lk = &ldsK [nb2][tid * 32];
        unsigned short* lv = &ldsVT[nb2][tid * 32];
        #pragma unroll
        for (int g = 0; g < 4; ++g) {
            async_cp16(gk + g * 8, lk + g * 8);
            async_cp16(gv + g * 8, lv + g * 8);
        }
    };
    stage(0, 0);    // prologue
#else
    {   // prologue: register-mediated copy of tile 0
        const uint4* gk = (const uint4*)(KgrB);
        const uint4* gv = (const uint4*)(VTgrB);
        uint4* lk = (uint4*)&ldsK [0][tid * 32];
        uint4* lv = (uint4*)&ldsVT[0][tid * 32];
        #pragma unroll
        for (int g = 0; g < 4; ++g) { lk[g] = gk[tid * 4 + g]; lv[g] = gv[tid * 4 + g]; }
    }
#endif

    for (int it = 0; it < NT; ++it) {
        const int buf = it & 1;
#if USE_ASYNC_LDS
        __builtin_amdgcn_s_wait_asynccnt(0);   // our async copies landed
#endif
        __syncthreads();                       // tile `it` visible; buf^1 free
        const bool have = (it + 1) < NT;
#if USE_ASYNC_LDS
        if (have) stage(buf ^ 1, it + 1);      // fire-and-forget next tile
#else
        uint4 pk[4], pv[4];
        if (have) {
            const uint4* gk = (const uint4*)(KgrB  + (size_t)(it + 1) * 8192 + tid * 32);
            const uint4* gv = (const uint4*)(VTgrB + (size_t)(it + 1) * 8192 + tid * 32);
            #pragma unroll
            for (int g = 0; g < 4; ++g) { pk[g] = gk[g]; pv[g] = gv[g]; }
        }
#endif
        // per-wave 32-bit key-mask (uniform -> SGPR)
        const unsigned kb = ballot32(kmb[it * 32 + lane] != 0);

        // ---- scores S^T = K_tile * Q^T, manually double-buffered fragments
        v8f T0 = {}, T1 = {};
        {
            v8u a0, a1, b0, b1;
            ldKfrag(buf, 0, 0, a0);
            ldKfrag(buf, 0, 1, b0);
            #pragma unroll
            for (int c = 0; c < 8; ++c) {
                if (c < 7) {
                    v8u& na = (c & 1) ? a0 : a1;
                    v8u& nb = (c & 1) ? b0 : b1;
                    ldKfrag(buf, c + 1, 0, na);
                    ldKfrag(buf, c + 1, 1, nb);
                }
                const v8u& ca = (c & 1) ? a1 : a0;
                const v8u& cb = (c & 1) ? b1 : b0;
                T0 = __builtin_amdgcn_wmma_f32_16x16x32_bf16(false, __builtin_bit_cast(v16bf, ca),
                                                             false, qf[c], (short)0, T0, false, false);
                T1 = __builtin_amdgcn_wmma_f32_16x16x32_bf16(false, __builtin_bit_cast(v16bf, cb),
                                                             false, qf[c], (short)0, T1, false, false);
            }
        }

        // ---- faithful masking + per-lane online softmax
        const unsigned kbl = kb >> (hl * 8);
        float s0v[8], s1v[8];
        float mloc = -3.0e38f;
        #pragma unroll
        for (int r = 0; r < 8; ++r) {
            float a  = T0[r]; if ((kbl >> r) & 1u)        a  = MASK_FILL;
            float c2 = T1[r]; if ((kbl >> (16 + r)) & 1u) c2 = MASK_FILL;
            s0v[r] = a; s1v[r] = c2;
            mloc = fmaxf(mloc, fmaxf(a, c2));
        }
        mloc = fmaxf(mloc, swapx16f(mloc));

        const bool need = ballot32(mloc > mrun) != 0u;
        float cscale = 1.0f;
        if (need) {
            const float mnew = fmaxf(mrun, mloc);
            cscale = __expf(mrun - mnew);
            mrun = mnew;
            #pragma unroll
            for (int cd = 0; cd < 16; ++cd)
                #pragma unroll
                for (int r = 0; r < 8; ++r) acc[cd][r] *= cscale;
        }

        float psum = 0.0f;
        #pragma unroll
        for (int r = 0; r < 8; ++r) {
            s0v[r] = __expf(s0v[r] - mrun);
            s1v[r] = __expf(s1v[r] - mrun);
            psum += s0v[r] + s1v[r];
        }
        lrun = lrun * cscale + psum + swapx16f(psum);

        // ---- build P^T B-fragment
        unsigned pa[4], pb[4];
        #pragma unroll
        for (int j = 0; j < 4; ++j) {
            pa[j] = bf2(s0v[2 * j], s0v[2 * j + 1]);
            pb[j] = bf2(s1v[2 * j], s1v[2 * j + 1]);
        }
        v8u pf;
        #pragma unroll
        for (int j = 0; j < 4; ++j) {
            unsigned ax = swapx16u(pa[j]);
            unsigned bx = swapx16u(pb[j]);
            pf[j]     = pa[j];
            pf[4 + j] = pb[j];
        }
        v16bf Pfrag = __builtin_bit_cast(v16bf, pf);

        // ---- O^T += V^T * P^T, manually double-buffered fragments
        {
            v8u uv0, uv1;
            ldVfrag(buf, 0, uv0);
            #pragma unroll
            for (int cd = 0; cd < 16; ++cd) {
                if (cd < 15) {
                    v8u& nu = (cd & 1) ? uv0 : uv1;
                    ldVfrag(buf, cd + 1, nu);
                }
                const v8u& cu = (cd & 1) ? uv1 : uv0;
                acc[cd] = __builtin_amdgcn_wmma_f32_16x16x32_bf16(false, __builtin_bit_cast(v16bf, cu),
                                                                  false, Pfrag, (short)0, acc[cd],
                                                                  false, false);
            }
        }

#if !USE_ASYNC_LDS
        if (have) {
            uint4* lk = (uint4*)&ldsK [buf ^ 1][tid * 32];
            uint4* lv = (uint4*)&ldsVT[buf ^ 1][tid * 32];
            #pragma unroll
            for (int g = 0; g < 4; ++g) { lk[g] = pk[g]; lv[g] = pv[g]; }
        }
#endif
    }

    // ---- epilogue
    const float wgt = qmb[qrow] ? 0.0f : (1.0f / lrun);
    #pragma unroll
    for (int cd = 0; cd < 16; ++cd) {
        float4 o0, o1;
        o0.x = acc[cd][0] * wgt; o0.y = acc[cd][1] * wgt;
        o0.z = acc[cd][2] * wgt; o0.w = acc[cd][3] * wgt;
        o1.x = acc[cd][4] * wgt; o1.y = acc[cd][5] * wgt;
        o1.z = acc[cd][6] * wgt; o1.w = acc[cd][7] * wgt;
        float* dst = Ob + (size_t)qrow * D + cd * 16 + hl * 8;
        *(float4*)(dst)     = o0;
        *(float4*)(dst + 4) = o1;
    }
}

// =====================================================================
// Fallback path (no scratch): staging + conversion in-loop.
// =====================================================================
__global__ __launch_bounds__(256)
void fa_bidir_kernel(const float* __restrict__ Q, const float* __restrict__ KV,
                     const unsigned char* __restrict__ qmask,
                     const unsigned char* __restrict__ kmask,
                     float* __restrict__ Out)
{
    constexpr int L = 1024, D = 256, NT = L / 32;
    const int b    = blockIdx.y;
    const int qt   = blockIdx.x;
    const int tid  = threadIdx.x;
    const int lane = tid & 31;
    const int wave = tid >> 5;
    const int hl   = lane >> 4;
    const int q16  = lane & 15;

    const float*         Qb  = Q  + (size_t)b * L * D;
    const float*         Kb  = KV + (size_t)b * L * D;
    const unsigned char* qmb = qmask + (size_t)b * L;
    const unsigned char* kmb = kmask + (size_t)b * L;
    float*               Ob  = Out + (size_t)b * L * D;

    __shared__ __align__(16) unsigned short ldsK [2][8192];
    __shared__ __align__(16) unsigned short ldsVT[2][8192];

    const int qrow = qt * 128 + wave * 16 + q16;
    const int k0   = (tid & 15) * 2;
    const int dgrp = tid >> 4;
    const int d0   = dgrp * 16;
    const int c0   = dgrp >> 1;
    const int s0   = (dgrp & 1) * 2;

    auto stage_store = [&](int nb2, const float4* fa, const float4* fb) {
        unsigned kA[8], kB[8];
        #pragma unroll
        for (int j = 0; j < 4; ++j) {
            kA[2 * j]     = bf2(fa[j].x, fa[j].y);
            kA[2 * j + 1] = bf2(fa[j].z, fa[j].w);
            kB[2 * j]     = bf2(fb[j].x, fb[j].y);
            kB[2 * j + 1] = bf2(fb[j].z, fb[j].w);
        }
        *(uint4*)&ldsK[nb2][KIDX(c0, s0,     k0)]     = make_uint4(kA[0], kA[1], kA[2], kA[3]);
        *(uint4*)&ldsK[nb2][KIDX(c0, s0 + 1, k0)]     = make_uint4(kA[4], kA[5], kA[6], kA[7]);
        *(uint4*)&ldsK[nb2][KIDX(c0, s0,     k0 + 1)] = make_uint4(kB[0], kB[1], kB[2], kB[3]);
        *(uint4*)&ldsK[nb2][KIDX(c0, s0 + 1, k0 + 1)] = make_uint4(kB[4], kB[5], kB[6], kB[7]);
        const int sv = k0 >> 3, ko = k0 & 7;
        #pragma unroll
        for (int i = 0; i < 8; ++i) {
            const int d = d0 + 2 * i;
            *(unsigned*)&ldsVT[nb2][VIDX(sv, d)     + ko] =
                __builtin_amdgcn_perm(kB[i], kA[i], 0x05040100u);
            *(unsigned*)&ldsVT[nb2][VIDX(sv, d + 1) + ko] =
                __builtin_amdgcn_perm(kB[i], kA[i], 0x07060302u);
        }
    };

    v16bf qf[8];
    #pragma unroll
    for (int c = 0; c < 8; ++c) {
        const float4* p = (const float4*)(Qb + (size_t)qrow * D + c * 32 + hl * 8);
        float4 f0 = p[0], f1 = p[1], f2 = p[4], f3 = p[5];
        v8u u;
        u[0] = bf2(f0.x, f0.y); u[1] = bf2(f0.z, f0.w);
        u[2] = bf2(f1.x, f1.y); u[3] = bf2(f1.z, f1.w);
        u[4] = bf2(f2.x, f2.y); u[5] = bf2(f2.z, f2.w);
        u[6] = bf2(f3.x, f3.y); u[7] = bf2(f3.z, f3.w);
        qf[c] = __builtin_bit_cast(v16bf, u);
    }

    v8f acc[16];
    { v8f z = {};
      #pragma unroll
      for (int i = 0; i < 16; ++i) acc[i] = z; }
    float mrun = -3.0e38f, lrun = 0.0f;

    {
        float4 fa[4], fb[4];
        const float4* sA = (const float4*)(Kb + (size_t)k0 * D + d0);
        const float4* sB = (const float4*)(Kb + (size_t)(k0 + 1) * D + d0);
        #pragma unroll
        for (int j = 0; j < 4; ++j) { fa[j] = sA[j]; fb[j] = sB[j]; }
        stage_store(0, fa, fb);
    }

    for (int it = 0; it < NT; ++it) {
        const int buf = it & 1;
        __syncthreads();

        float4 na[4], nb4[4];
        const bool have = (it + 1) < NT;
        if (have) {
            const size_t base = (size_t)((it + 1) * 32) * D;
            const float4* sA = (const float4*)(Kb + base + (size_t)k0 * D + d0);
            const float4* sB = (const float4*)(Kb + base + (size_t)(k0 + 1) * D + d0);
            #pragma unroll
            for (int j = 0; j < 4; ++j) { na[j] = sA[j]; nb4[j] = sB[j]; }
        }
        const unsigned kb = ballot32(kmb[it * 32 + lane] != 0);

        v8f T0 = {}, T1 = {};
        #pragma unroll
        for (int c = 0; c < 8; ++c) {
            v8u ua, ub;
            { uint4 x = *(const uint4*)&ldsK[buf][KIDX(c, hl,     q16)];
              ua[0]=x.x; ua[1]=x.y; ua[2]=x.z; ua[3]=x.w; }
            { uint4 x = *(const uint4*)&ldsK[buf][KIDX(c, 2 + hl, q16)];
              ua[4]=x.x; ua[5]=x.y; ua[6]=x.z; ua[7]=x.w; }
            { uint4 x = *(const uint4*)&ldsK[buf][KIDX(c, hl,     16 + q16)];
              ub[0]=x.x; ub[1]=x.y; ub[2]=x.z; ub[3]=x.w; }
            { uint4 x = *(const uint4*)&ldsK[buf][KIDX(c, 2 + hl, 16 + q16)];
              ub[4]=x.x; ub[5]=x.y; ub[6]=x.z; ub[7]=x.w; }
            T0 = __builtin_amdgcn_wmma_f32_16x16x32_bf16(false, __builtin_bit_cast(v16bf, ua),
                                                         false, qf[c], (short)0, T0, false, false);
            T1 = __builtin_amdgcn_wmma_f32_16x16x32_bf16(false, __builtin_bit_cast(v16bf, ub),
                                                         false, qf[c], (short)0, T1, false, false);
        }

        const unsigned kbl = kb >> (hl * 8);
        float s0v[8], s1v[8];
        float mloc = -3.0e38f;
        #pragma unroll
        for (int r = 0; r < 8; ++r) {
            float a  = T0[r]; if ((kbl >> r) & 1u)        a  = MASK_FILL;
            float c2 = T1[r]; if ((kbl >> (16 + r)) & 1u) c2 = MASK_FILL;
            s0v[r] = a; s1v[r] = c2;
            mloc = fmaxf(mloc, fmaxf(a, c2));
        }
        mloc = fmaxf(mloc, swapx16f(mloc));

        const bool need = ballot32(mloc > mrun) != 0u;
        float cscale = 1.0f;
        if (need) {
            const float mnew = fmaxf(mrun, mloc);
            cscale = __expf(mrun - mnew);
            mrun = mnew;
            #pragma unroll
            for (int cd = 0; cd < 16; ++cd)
                #pragma unroll
                for (int r = 0; r < 8; ++r) acc[cd][r] *= cscale;
        }

        float psum = 0.0f;
        #pragma unroll
        for (int r = 0; r < 8; ++r) {
            s0v[r] = __expf(s0v[r] - mrun);
            s1v[r] = __expf(s1v[r] - mrun);
            psum += s0v[r] + s1v[r];
        }
        lrun = lrun * cscale + psum + swapx16f(psum);

        unsigned pa[4], pb[4];
        #pragma unroll
        for (int j = 0; j < 4; ++j) {
            pa[j] = bf2(s0v[2 * j], s0v[2 * j + 1]);
            pb[j] = bf2(s1v[2 * j], s1v[2 * j + 1]);
        }
        v8u pf;
        #pragma unroll
        for (int j = 0; j < 4; ++j) {
            unsigned ax = swapx16u(pa[j]);
            unsigned bx = swapx16u(pb[j]);
            pf[j]     = pa[j];
            pf[4 + j] = pb[j];
        }
        v16bf Pfrag = __builtin_bit_cast(v16bf, pf);

        #pragma unroll
        for (int cd = 0; cd < 16; ++cd) {
            v8u uv;
            { uint4 x = *(const uint4*)&ldsVT[buf][VIDX(hl,     cd * 16 + q16)];
              uv[0]=x.x; uv[1]=x.y; uv[2]=x.z; uv[3]=x.w; }
            { uint4 x = *(const uint4*)&ldsVT[buf][VIDX(2 + hl, cd * 16 + q16)];
              uv[4]=x.x; uv[5]=x.y; uv[6]=x.z; uv[7]=x.w; }
            acc[cd] = __builtin_amdgcn_wmma_f32_16x16x32_bf16(false, __builtin_bit_cast(v16bf, uv),
                                                              false, Pfrag, (short)0, acc[cd],
                                                              false, false);
        }

        if (have) stage_store(buf ^ 1, na, nb4);
    }

    const float wgt = qmb[qrow] ? 0.0f : (1.0f / lrun);
    #pragma unroll
    for (int cd = 0; cd < 16; ++cd) {
        float4 o0, o1;
        o0.x = acc[cd][0] * wgt; o0.y = acc[cd][1] * wgt;
        o0.z = acc[cd][2] * wgt; o0.w = acc[cd][3] * wgt;
        o1.x = acc[cd][4] * wgt; o1.y = acc[cd][5] * wgt;
        o1.z = acc[cd][6] * wgt; o1.w = acc[cd][7] * wgt;
        float* dst = Ob + (size_t)qrow * D + cd * 16 + hl * 8;
        *(float4*)(dst)     = o0;
        *(float4*)(dst + 4) = o1;
    }
}

extern "C" void kernel_launch(void* const* d_in, const int* in_sizes, int n_in,
                              void* d_out, int out_size, void* d_ws, size_t ws_size,
                              hipStream_t stream) {
    const float*         v1  = (const float*)d_in[0];
    const unsigned char* v1m = (const unsigned char*)d_in[1];
    const float*         v2  = (const float*)d_in[2];
    const unsigned char* v2m = (const unsigned char*)d_in[3];

    float* att1 = (float*)d_out;
    float* att2 = att1 + (size_t)64 * 1024 * 256;

    dim3 blk(256), grid(8, 64);
    const size_t LAYOUT = 16777216ull;            // ushorts per 32MB layout
    if (ws_size >= 4ull * LAYOUT * 2ull) {        // 128 MB scratch available
        unsigned short* wsu = (unsigned short*)d_ws;
        unsigned short* K1 = wsu;
        unsigned short* V1 = wsu + LAYOUT;
        unsigned short* K2 = wsu + 2 * LAYOUT;
        unsigned short* V2 = wsu + 3 * LAYOUT;
        dim3 gp(32, 64);
        pack_bf16_kernel<<<gp, blk, 0, stream>>>(v1, K1, V1);
        pack_bf16_kernel<<<gp, blk, 0, stream>>>(v2, K2, V2);
        fa_pre_kernel<<<grid, blk, 0, stream>>>(v1, K2, V2, v1m, v2m, att1);
        fa_pre_kernel<<<grid, blk, 0, stream>>>(v2, K1, V1, v2m, v1m, att2);
    } else {
        fa_bidir_kernel<<<grid, blk, 0, stream>>>(v1, v2, v1m, v2m, att1);
        fa_bidir_kernel<<<grid, blk, 0, stream>>>(v2, v1, v2m, v1m, att2);
    }
}
